// Attention_403726926044
// MI455X (gfx1250) — matrix operand order repaired and checked by reading, not compiled, not run on hardware
//
#include <hip/hip_runtime.h>
#include <hip/hip_bf16.h>

// ---------------------------------------------------------------------------
// Multi-query attention for MI455X (gfx1250), bf16 WMMA pipeline.
//   inputs (setup_inputs order): x, pos_bias, gamma, wq, wk, wv, wo, mask
//   pipeline:
//     [1] transpose-cast weights fp32 -> bf16  (B^T layout: [N][K])
//     [2] RMSNorm(x)*gamma -> bf16 xn
//     [3] WMMA GEMM xn @ wq/wk/wv  (q pre-scaled, v stored transposed)
//     [4] flash attention: QK^T + pos_bias + causal(1e-10) online-softmax, P@V
//     [5] WMMA GEMM attn_out @ wo -> fp32 d_out
// All matrix math uses v_wmma_f32_16x16x32_bf16 (f32 accumulate).
// The K-loop is manually double-buffered and fenced with
// __builtin_amdgcn_sched_barrier(0) so the machine scheduler cannot sink the
// global_load_b128 clauses back under their consuming WMMAs (which produced
// a full s_wait_loadcnt 0 before every v_wmma in earlier builds).
// ---------------------------------------------------------------------------

#define DIMV      2048
#define SEQV      2048
#define HEADSV    16
#define DHEADV    128
#define EPSV      1e-5f
#define MASKVAL   1e-10f
#define SCALEV    0.08838834764831845f   // 128^-0.5

typedef __bf16 bf16;
typedef __attribute__((ext_vector_type(16))) __bf16 v16bf;
typedef __attribute__((ext_vector_type(8)))  __bf16 v8bf;
typedef __attribute__((ext_vector_type(8)))  float  v8f;

union Frag16 { v16bf v; v8bf h[2]; };

__device__ __forceinline__ v8f zero8() {
  v8f z;
  #pragma unroll
  for (int i = 0; i < 8; ++i) z[i] = 0.0f;
  return z;
}

__device__ __forceinline__ v8f wmma_bf16(v16bf a, v16bf b, v8f c) {
  // D = A(16x32) * B(32x16) + C(16x16 f32)
  return __builtin_amdgcn_wmma_f32_16x16x32_bf16(
      /*neg_a=*/false, a, /*neg_b=*/false, b,
      /*c_mod=*/(short)0, c, /*reuse_a=*/false, /*reuse_b=*/false);
}

// A fragment 16x32 (MxK), A row-major with leading dim ld.
// ISA layout: lanes 0-15 row M=lane hold K = {0..7, 16..23};
//             lanes 16-31 row M=lane-16 hold K = {8..15, 24..31}.
__device__ __forceinline__ v16bf load_a_frag(const bf16* base, int ld) {
  int lane = threadIdx.x & 31;
  int row  = lane & 15;
  int hi   = lane >> 4;
  const bf16* p = base + (size_t)row * ld;
  Frag16 f;
  f.h[0] = *(const v8bf*)(p + 8 * hi);        // K = 8*hi .. 8*hi+7
  f.h[1] = *(const v8bf*)(p + 16 + 8 * hi);   // K = 16+8*hi .. 16+8*hi+7
  return f.v;
}

// B fragment 32x16 (KxN), sourced from B^T [N][K] row-major with leading dim ld.
// ISA layout: lanes 0-15 col N=lane hold K = {0..7, 16..23}; lanes 16-31 hold K = {8..15, 24..31}.
__device__ __forceinline__ v16bf load_bt_frag(const bf16* base, int ld) {
  int lane = threadIdx.x & 31;
  int col  = lane & 15;
  int hi   = lane >> 4;
  const bf16* p = base + (size_t)col * ld;
  Frag16 f;
  f.h[0] = *(const v8bf*)(p + 8 * hi);
  f.h[1] = *(const v8bf*)(p + 16 + 8 * hi);
  return f.v;
}

// ---------------------------------------------------------------------------
// [1] transpose + cast: src f32 [R][C] row-major -> dst bf16 [C][R] row-major
// ---------------------------------------------------------------------------
__global__ __launch_bounds__(256) void transpose_cast_kernel(
    const float* __restrict__ src, bf16* __restrict__ dst, int R, int C) {
  __shared__ float tile[32][33];
  int tx = threadIdx.x, ty = threadIdx.y;     // 32 x 8
  int c0 = blockIdx.x * 32, r0 = blockIdx.y * 32;
  #pragma unroll
  for (int rr = ty; rr < 32; rr += 8)
    tile[rr][tx] = src[(size_t)(r0 + rr) * C + (c0 + tx)];
  __syncthreads();
  #pragma unroll
  for (int rr = ty; rr < 32; rr += 8)
    dst[(size_t)(c0 + rr) * R + (r0 + tx)] = (bf16)tile[tx][rr];
}

// ---------------------------------------------------------------------------
// [2] RMSNorm: xn = x * rsqrt(mean(x^2)+eps) * gamma, cast bf16. 1 block/row.
// ---------------------------------------------------------------------------
__global__ __launch_bounds__(256) void rmsnorm_kernel(
    const float* __restrict__ x, const float* __restrict__ gamma,
    bf16* __restrict__ xn) {
  __shared__ float red[256];
  int row = blockIdx.x;
  const float* xr = x + (size_t)row * DIMV;
  float s = 0.0f;
  for (int i = threadIdx.x; i < DIMV; i += 256) { float v = xr[i]; s += v * v; }
  red[threadIdx.x] = s;
  __syncthreads();
  #pragma unroll
  for (int o = 128; o > 0; o >>= 1) {
    if ((int)threadIdx.x < o) red[threadIdx.x] += red[threadIdx.x + o];
    __syncthreads();
  }
  float inv = rsqrtf(red[0] * (1.0f / DIMV) + EPSV);
  for (int i = threadIdx.x; i < DIMV; i += 256)
    xn[(size_t)row * DIMV + i] = (bf16)(xr[i] * inv * gamma[i]);
}

// ---------------------------------------------------------------------------
// [3]/[5] GEMM: C[M][N] = A[M][K](bf16,row) @ B (via B^T[N][K] bf16,row)
// One wave computes a 16x64 tile. K-loop is manually 2x unrolled with
// rotating (a0,b0)/(a1,b1) buffers; sched_barrier(0) pins each load batch
// ahead of the opposite buffer's WMMA chain, so every WMMA group consumes
// fragments loaded one half-iteration earlier (loads fly under matrix ops).
// Requires K % 64 == 0 (true for all call sites: K = 2048).
// mode 0: bf16 row-major C             (k projection -> [S][128])
// mode 1: bf16 scattered q layout      q[h][i][d], scaled by SCALE
// mode 2: bf16 transposed C            v_t[d][i]
// mode 3: f32 row-major C              (final output)
// ---------------------------------------------------------------------------
__global__ __launch_bounds__(256) void gemm_bt_kernel(
    const bf16* __restrict__ A, const bf16* __restrict__ BT,
    void* __restrict__ Cout, int M, int N, int K, float scale, int mode) {
  int wave = threadIdx.x >> 5;
  int lane = threadIdx.x & 31;
  int hi = lane >> 4, ln = lane & 15;
  int w = blockIdx.x * 8 + wave;
  int tilesN = N >> 6;
  int m0 = (w / tilesN) * 16;
  int n0 = (w % tilesN) * 64;

  v8f c[4];
  #pragma unroll
  for (int t = 0; t < 4; ++t) c[t] = zero8();

  const bf16* Ab = A + (size_t)m0 * K;
  const bf16* Bb = BT + (size_t)n0 * K;

  // prologue: buffer 0 holds K-step 0
  v16bf a0 = load_a_frag(Ab, K);
  v16bf b0[4];
  #pragma unroll
  for (int t = 0; t < 4; ++t) b0[t] = load_bt_frag(Bb + (size_t)(16 * t) * K, K);

  for (int k0 = 0; k0 < K; k0 += 64) {
    // issue buffer-1 loads (K-step k0+32); always valid since K % 64 == 0
    v16bf a1 = load_a_frag(Ab + k0 + 32, K);
    v16bf b1[4];
    #pragma unroll
    for (int t = 0; t < 4; ++t)
      b1[t] = load_bt_frag(Bb + (size_t)(16 * t) * K + k0 + 32, K);
    __builtin_amdgcn_sched_barrier(0);

    // consume buffer 0 (its loads have been in flight a half-iteration)
    #pragma unroll
    for (int t = 0; t < 4; ++t) c[t] = wmma_bf16(a0, b0[t], c[t]);

    // refill buffer 0 for K-step k0+64 while buffer-1 WMMAs run next
    if (k0 + 64 < K) {
      a0 = load_a_frag(Ab + k0 + 64, K);
      #pragma unroll
      for (int t = 0; t < 4; ++t)
        b0[t] = load_bt_frag(Bb + (size_t)(16 * t) * K + k0 + 64, K);
      __builtin_prefetch(Ab + (size_t)ln * K + k0 + 128, 0, 1);
    }
    __builtin_amdgcn_sched_barrier(0);

    // consume buffer 1
    #pragma unroll
    for (int t = 0; t < 4; ++t) c[t] = wmma_bf16(a1, b1[t], c[t]);
  }

  // C layout: element r -> row m0 + r + 8*hi, col n0 + 16*t + ln
  #pragma unroll
  for (int t = 0; t < 4; ++t) {
    #pragma unroll
    for (int r = 0; r < 8; ++r) {
      int i   = m0 + r + 8 * hi;
      int col = n0 + 16 * t + ln;
      float v = c[t][r] * scale;
      if (mode == 0) {
        ((bf16*)Cout)[(size_t)i * N + col] = (bf16)v;
      } else if (mode == 1) {          // q[h][i][d], h=col>>7, d=col&127
        ((bf16*)Cout)[((size_t)(col >> 7) * M + i) * DHEADV + (col & 127)] = (bf16)v;
      } else if (mode == 2) {          // v_t[col][i]
        ((bf16*)Cout)[(size_t)col * M + i] = (bf16)v;
      } else {
        ((float*)Cout)[(size_t)i * N + col] = v;
      }
    }
  }
}

// ---------------------------------------------------------------------------
// [4] flash attention. 1 wave per (head, 16-row q tile); 4 waves / block.
//   q  : [H][S][128] bf16 (pre-scaled)     -> A frags (live in regs all loop)
//   kb : [S][128]    bf16 (== B^T of K^T)  -> B frags for QK^T
//   vt : [128][S]    bf16 (== B^T of V)    -> B frags for P@V
//   bias: [H][S][S]  f32, streamed once (dominant memory traffic, 256 MB)
//   masked positions get value 1e-10 (faithful), so all columns participate.
// sched_barrier(0) pins the bias+K load batch ahead of the QK^T WMMAs and
// the V load batch ahead of the softmax VALU block (whose exp/shuffle work
// hides the V latency before the P@V WMMAs).
// ---------------------------------------------------------------------------
__global__ __launch_bounds__(128) void flash_attn_kernel(
    const bf16* __restrict__ qh, const bf16* __restrict__ kb,
    const bf16* __restrict__ vt, const float* __restrict__ pos_bias,
    bf16* __restrict__ ao) {
  __shared__ bf16 pbuf[4][16 * 40];            // per-wave 16x32 P tile, padded
  int wave = threadIdx.x >> 5;
  int lane = threadIdx.x & 31;
  int hi = lane >> 4, ln = lane & 15;
  int h  = blockIdx.y;
  int i0 = (blockIdx.x * 4 + wave) * 16;

  // Q A-frags: 16x128 = 4 frags of 16x32
  v16bf qa[4];
  const bf16* qbase = qh + ((size_t)h * SEQV + i0) * DHEADV;
  #pragma unroll
  for (int kk = 0; kk < 4; ++kk) qa[kk] = load_a_frag(qbase + kk * 32, DHEADV);

  v8f acc[8];
  #pragma unroll
  for (int t = 0; t < 8; ++t) acc[t] = zero8();
  float rowmax[8], rowsum[8];
  #pragma unroll
  for (int r = 0; r < 8; ++r) { rowmax[r] = -1e30f; rowsum[r] = 0.0f; }

  const float* bias = pos_bias + (size_t)h * SEQV * SEQV;
  bf16* pb = pbuf[wave];

  for (int j0 = 0; j0 < SEQV; j0 += 32) {
    // issue the 16 pos_bias loads first; they complete under the WMMA chain
    float ba[8], bb[8];
    #pragma unroll
    for (int r = 0; r < 8; ++r) {
      int i = i0 + r + 8 * hi;
      ba[r] = bias[(size_t)i * SEQV + j0 + ln];
      bb[r] = bias[(size_t)i * SEQV + j0 + 16 + ln];
    }

    // batch all 8 K B-frags, then fence, then run the 8 QK^T WMMAs
    v16bf kf[8];
    #pragma unroll
    for (int kk = 0; kk < 4; ++kk) {
      kf[kk]     = load_bt_frag(kb + (size_t)j0 * DHEADV + kk * 32, DHEADV);
      kf[4 + kk] = load_bt_frag(kb + (size_t)(j0 + 16) * DHEADV + kk * 32, DHEADV);
    }
    __builtin_amdgcn_sched_barrier(0);
    v8f s0 = zero8(), s1 = zero8();
    #pragma unroll
    for (int kk = 0; kk < 4; ++kk) s0 = wmma_bf16(qa[kk], kf[kk], s0);
    #pragma unroll
    for (int kk = 0; kk < 4; ++kk) s1 = wmma_bf16(qa[kk], kf[4 + kk], s1);

    // batch the 8 V B-frags now: latency hides under softmax VALU work below
    v16bf vf[8];
    #pragma unroll
    for (int t = 0; t < 8; ++t)
      vf[t] = load_bt_frag(vt + (size_t)(t * 16) * SEQV + j0, SEQV);
    __builtin_amdgcn_sched_barrier(0);

    // + pos_bias, causal mask -> MASKVAL (not -inf; still contributes)
    #pragma unroll
    for (int r = 0; r < 8; ++r) {
      int i  = i0 + r + 8 * hi;
      int ja = j0 + ln, jb = j0 + 16 + ln;
      float sa = s0[r] + ba[r];
      float sb = s1[r] + bb[r];
      s0[r] = (ja <= i) ? sa : MASKVAL;
      s1[r] = (jb <= i) ? sb : MASKVAL;
    }

    // online softmax per row (16 lanes share a row; xor masks stay in-half)
    float alpha[8];
    #pragma unroll
    for (int r = 0; r < 8; ++r) {
      float m = fmaxf(s0[r], s1[r]);
      #pragma unroll
      for (int o = 8; o >= 1; o >>= 1) m = fmaxf(m, __shfl_xor(m, o, 32));
      float mnew = fmaxf(rowmax[r], m);
      float al = __expf(rowmax[r] - mnew);
      float p0 = __expf(s0[r] - mnew);
      float p1 = __expf(s1[r] - mnew);
      float ps = p0 + p1;
      #pragma unroll
      for (int o = 8; o >= 1; o >>= 1) ps += __shfl_xor(ps, o, 32);
      rowsum[r] = rowsum[r] * al + ps;
      rowmax[r] = mnew;
      alpha[r] = al;
      s0[r] = p0; s1[r] = p1;
    }
    #pragma unroll
    for (int t = 0; t < 8; ++t)
      #pragma unroll
      for (int r = 0; r < 8; ++r) acc[t][r] *= alpha[r];

    // re-stripe P: C layout -> A layout via per-wave LDS (DS in-order per wave)
    #pragma unroll
    for (int r = 0; r < 8; ++r) {
      int row = r + 8 * hi;
      pb[row * 40 + ln]      = (bf16)s0[r];
      pb[row * 40 + 16 + ln] = (bf16)s1[r];
    }
    __asm__ volatile("" ::: "memory");
    v16bf pa = load_a_frag(pb, 40);

    // acc += P(16x32) @ V(32x128), 8 N-tiles, all B-frags already resident
    #pragma unroll
    for (int t = 0; t < 8; ++t)
      acc[t] = wmma_bf16(pa, vf[t], acc[t]);
  }

  // normalize and store as [S][H*128] bf16 for the output projection
  #pragma unroll
  for (int t = 0; t < 8; ++t) {
    #pragma unroll
    for (int r = 0; r < 8; ++r) {
      int i = i0 + r + 8 * hi;
      int d = t * 16 + ln;
      ao[(size_t)i * (HEADSV * DHEADV) + h * DHEADV + d] =
          (bf16)(acc[t][r] / rowsum[r]);
    }
  }
}

// ---------------------------------------------------------------------------
// launch
// ---------------------------------------------------------------------------
extern "C" void kernel_launch(void* const* d_in, const int* in_sizes, int n_in,
                              void* d_out, int out_size, void* d_ws, size_t ws_size,
                              hipStream_t stream) {
  const float* x        = (const float*)d_in[0];
  const float* pos_bias = (const float*)d_in[1];
  const float* gamma    = (const float*)d_in[2];
  const float* wq       = (const float*)d_in[3];
  const float* wk       = (const float*)d_in[4];
  const float* wv       = (const float*)d_in[5];
  const float* wo       = (const float*)d_in[6];
  (void)in_sizes; (void)n_in; (void)out_size; (void)ws_size;
  // mask (d_in[7]) is the causal tril; computed analytically in-kernel.

  // workspace layout (bytes); total ~44 MB
  char* ws = (char*)d_ws;
  size_t off = 0;
  bf16* xn   = (bf16*)(ws + off); off += (size_t)SEQV * DIMV * 2;            // 8 MB
  bf16* wq_t = (bf16*)(ws + off); off += (size_t)DIMV * DIMV * 2;            // 8 MB
  bf16* wk_t = (bf16*)(ws + off); off += (size_t)DHEADV * DIMV * 2;          // 0.5 MB
  bf16* wv_t = (bf16*)(ws + off); off += (size_t)DHEADV * DIMV * 2;          // 0.5 MB
  bf16* wo_t = (bf16*)(ws + off); off += (size_t)DIMV * DIMV * 2;            // 8 MB
  bf16* q_h  = (bf16*)(ws + off); off += (size_t)HEADSV * SEQV * DHEADV * 2; // 8 MB
  bf16* k_b  = (bf16*)(ws + off); off += (size_t)SEQV * DHEADV * 2;          // 0.5 MB
  bf16* v_t  = (bf16*)(ws + off); off += (size_t)DHEADV * SEQV * 2;          // 0.5 MB
  bf16* ao   = (bf16*)(ws + off); off += (size_t)SEQV * DIMV * 2;            // 8 MB

  dim3 tb(32, 8);
  // [1] weights -> bf16 B^T layouts
  transpose_cast_kernel<<<dim3(DIMV / 32, DIMV / 32),   tb, 0, stream>>>(wq, wq_t, DIMV, DIMV);
  transpose_cast_kernel<<<dim3(DHEADV / 32, DIMV / 32), tb, 0, stream>>>(wk, wk_t, DIMV, DHEADV);
  transpose_cast_kernel<<<dim3(DHEADV / 32, DIMV / 32), tb, 0, stream>>>(wv, wv_t, DIMV, DHEADV);
  transpose_cast_kernel<<<dim3(DIMV / 32, DIMV / 32),   tb, 0, stream>>>(wo, wo_t, DIMV, DIMV);

  // [2] RMSNorm
  rmsnorm_kernel<<<SEQV, 256, 0, stream>>>(x, gamma, xn);

  // [3] projections (waves = (M/16)*(N/64); 8 waves/block of 256 threads)
  gemm_bt_kernel<<<(SEQV / 16) * (DIMV / 64) / 8, 256, 0, stream>>>(
      xn, wq_t, q_h, SEQV, DIMV, DIMV, SCALEV, 1);
  gemm_bt_kernel<<<(SEQV / 16) * (DHEADV / 64) / 8, 256, 0, stream>>>(
      xn, wk_t, k_b, SEQV, DHEADV, DIMV, 1.0f, 0);
  gemm_bt_kernel<<<(SEQV / 16) * (DHEADV / 64) / 8, 256, 0, stream>>>(
      xn, wv_t, v_t, SEQV, DHEADV, DIMV, 1.0f, 2);

  // [4] flash attention: grid (qtiles/4, heads), 4 waves per block
  flash_attn_kernel<<<dim3(SEQV / 16 / 4, HEADSV), 128, 0, stream>>>(
      q_h, k_b, v_t, pos_bias, ao);

  // [5] output projection -> fp32 d_out
  gemm_bt_kernel<<<(SEQV / 16) * (DIMV / 64) / 8, 256, 0, stream>>>(
      ao, wo_t, d_out, SEQV, DIMV, DIMV, 1.0f, 3);
}
